// SpanSelector_49735721288073
// MI455X (gfx1250) — matrix operand order repaired and checked by reading, not compiled, not run on hardware
//
#include <hip/hip_runtime.h>
#include <stdint.h>
#include <math.h>

// ---------------------------------------------------------------------------
// SpanSelector for MI455X (gfx1250, wave32, WMMA).
// All GEMMs run on v_wmma_f32_16x16x32_bf16 (fp32 accumulate).
// ---------------------------------------------------------------------------

#define B_    64
#define T_    128
#define D_    768
#define H_    256
#define J_    128
#define BT_   (B_ * T_)          // 8192
#define NSPAN 8256               // T*(T+1)/2
#define KSEL  256                // 2*T
#define NTILE_SPAN (NSPAN / 16)  // 516 (exact)

typedef __attribute__((ext_vector_type(16))) __bf16 v16bf;
typedef __attribute__((ext_vector_type(8)))  __bf16 v8bf;
typedef __attribute__((ext_vector_type(8)))  float  v8f;

#if defined(__has_builtin)
#if __has_builtin(__builtin_amdgcn_sched_group_barrier)
#define SCHED_GROUP(mask, size, id) __builtin_amdgcn_sched_group_barrier((mask), (size), (id))
#endif
#endif
#ifndef SCHED_GROUP
#define SCHED_GROUP(mask, size, id)
#endif

// round-to-nearest-even f32 -> bf16 (as raw u16)
static __device__ inline uint16_t f2bf(float f) {
  uint32_t u = __float_as_uint(f);
  uint32_t r = u + 0x7FFFu + ((u >> 16) & 1u);
  return (uint16_t)(r >> 16);
}

// A-fragment (16x32 bf16, ISA 7.12.2 16-bit A layout):
// lane = {h=lane[4], row=lane[3:0]}; elems 0..7 = K 8h..8h+7, elems 8..15 = K 16+8h..
// rowp points at (row base + kBase) in a row-major bf16 array.
static __device__ inline v16bf load_afrag(const uint16_t* rowp, int h) {
  v8bf lo = *reinterpret_cast<const v8bf*>(rowp + 8 * h);
  v8bf hi = *reinterpret_cast<const v8bf*>(rowp + 16 + 8 * h);
  return __builtin_shufflevector(lo, hi, 0, 1, 2, 3, 4, 5, 6, 7,
                                 8, 9, 10, 11, 12, 13, 14, 15);
}

static __device__ inline v8f wmma_bf16(v16bf a, v16bf b, v8f c) {
  // (neg_a, A, neg_b, B, c_mod, C, reuse_a, reuse_b)
  return __builtin_amdgcn_wmma_f32_16x16x32_bf16(false, a, false, b,
                                                 (short)0, c, false, false);
}

// ---------------------------------------------------------------------------
// Prep kernels
// ---------------------------------------------------------------------------
__global__ void k_cvt_bf16(uint16_t* __restrict__ dst,
                           const float* __restrict__ src, int n) {
  int i = blockIdx.x * blockDim.x + threadIdx.x;
  if (i < n) dst[i] = f2bf(src[i]);
}

// Pack a row-major fp32 weight [K,N] (row stride 256) into the per-lane
// B-fragment layout: dst[((kb*NB+nb)*32+lane)*16 + e] where
// k = kb*32 + 8*(lane>>4) + (e&7) + 16*(e>>3), n = nb*16 + (lane&15).
// Columns >= split come from s1 (used to fuse W_start|W_end).
__global__ void k_pack_w(uint16_t* __restrict__ dst,
                         const float* __restrict__ s0,
                         const float* __restrict__ s1,
                         int NB, int split, int total) {
  int d = blockIdx.x * blockDim.x + threadIdx.x;
  if (d >= total) return;
  int e    = d & 15;
  int lane = (d >> 4) & 31;
  int tile = d >> 9;
  int nb   = tile % NB;
  int kb   = tile / NB;
  int k = kb * 32 + ((lane >> 4) << 3) + (e & 7) + ((e >> 3) << 4);
  int n = nb * 16 + (lane & 15);
  float v = (n < split) ? s0[k * 256 + n] : s1[k * 256 + (n - split)];
  dst[d] = f2bf(v);
}

__global__ void k_span_idx(int* __restrict__ sIdx, int* __restrict__ eIdx) {
  int n = blockIdx.x * blockDim.x + threadIdx.x;
  if (n >= NSPAN) return;
  int s = 0, base = 0;
  while (base + (T_ - s) <= n) { base += (T_ - s); ++s; }
  sIdx[n] = s;
  eIdx[n] = s + (n - base);
}

// ---------------------------------------------------------------------------
// K1: fused projection GEMM  AE[8192,512] = X[8192,768] @ [W_start|W_end] + bias
// One wave owns one 16-row strip and 8 N-tiles (8 accumulators). Per K-step:
// fetch the A-fragment once, batch-load the 8 B-fragments, and pin the
// schedule with sched_group_barrier so each WMMA keeps >=3 B-fragments
// (6 b128 loads) in flight -> partial s_wait_loadcnt instead of full drains.
// ---------------------------------------------------------------------------
__global__ void k_proj(const uint16_t* __restrict__ Xbf,
                       const uint16_t* __restrict__ Wse,
                       const float* __restrict__ b_start,
                       const float* __restrict__ b_end,
                       float* __restrict__ AE) {
  int wave = threadIdx.x >> 5;         // 4 waves/block
  int lane = threadIdx.x & 31;
  int h = lane >> 4, r = lane & 15;
  int mTile = blockIdx.x;              // 512 tiles of M
  int nt0   = wave * 8;                // this wave's 8 of 32 N-tiles
  const int NB = 32;
  v8f acc[8];
#pragma unroll
  for (int i = 0; i < 8; ++i) acc[i] = {};
  const uint16_t* arow = Xbf + (size_t)(mTile * 16 + r) * D_;
#pragma unroll 1
  for (int kb = 0; kb < 24; ++kb) {
    v16bf a = load_afrag(arow + kb * 32, h);
    const uint16_t* bbase = Wse + ((size_t)((kb * NB + nt0) * 32 + lane) << 4);
    v16bf bfrag[8];
#pragma unroll
    for (int i = 0; i < 8; ++i)
      bfrag[i] = *reinterpret_cast<const v16bf*>(bbase + (i << 9));  // 1 KB/tile
#pragma unroll
    for (int i = 0; i < 8; ++i)
      acc[i] = wmma_bf16(a, bfrag[i], acc[i]);
    // Schedule shape: [10 vmem] ([1 wmma][2 vmem])x4 [4 wmma]
    // -> every WMMA retains several B-fragments in flight.
    SCHED_GROUP(0x020, 10, 0);             // A-frag + B0..B3 loads
    SCHED_GROUP(0x008, 1, 0);
    SCHED_GROUP(0x020, 2, 0);              // B4
    SCHED_GROUP(0x008, 1, 0);
    SCHED_GROUP(0x020, 2, 0);              // B5
    SCHED_GROUP(0x008, 1, 0);
    SCHED_GROUP(0x020, 2, 0);              // B6
    SCHED_GROUP(0x008, 1, 0);
    SCHED_GROUP(0x020, 2, 0);              // B7
    SCHED_GROUP(0x008, 4, 0);              // drain
  }
#pragma unroll
  for (int i = 0; i < 8; ++i) {
    int col = (nt0 + i) * 16 + r;
    float bias = (col < H_) ? b_start[col] : b_end[col - H_];
    float* out = AE + (size_t)(mTile * 16 + 8 * h) * 512 + col;
#pragma unroll
    for (int j = 0; j < 8; ++j) out[(size_t)j * 512] = acc[i][j] + bias;
  }
}

// ---------------------------------------------------------------------------
// K2: span scorer. Block = 256 thr (8 waves) handles 16 spans of one batch:
//  phase1: act[16][256] = relu(A[s]+E[e]) -> bf16 in LDS
//  phase2: h = act @ W_s1 (+b_s1), relu, fold dot with W_s2 via shfl-reduce
//  phase3: masked score -> scores[b][n]
// Each wave owns 2 N-tiles; 8 waves/block provide the latency hiding here.
// ---------------------------------------------------------------------------
__global__ void k_scorer(const float* __restrict__ AE,
                         const int* __restrict__ sIdx,
                         const int* __restrict__ eIdx,
                         const int* __restrict__ mask,
                         const uint16_t* __restrict__ Ws1,
                         const float* __restrict__ b_s1,
                         const float* __restrict__ W_s2,
                         const float* __restrict__ b_s2,
                         float* __restrict__ scores) {
  __shared__ __align__(16) uint16_t actL[16 * 256];
  __shared__ float partL[16][16];
  int tile = blockIdx.x, b = blockIdx.y;
  int tid = threadIdx.x;
  {  // phase 1: build activation tile
    int r = tid >> 4, c = tid & 15;
    int n = tile * 16 + r;
    int s = sIdx[n], e = eIdx[n];
    const float* ar = AE + (size_t)(b * T_ + s) * 512 + c * 16;
    const float* er = AE + (size_t)(b * T_ + e) * 512 + H_ + c * 16;
#pragma unroll
    for (int q = 0; q < 4; ++q) {
      float4 av = *reinterpret_cast<const float4*>(ar + 4 * q);
      float4 ev = *reinterpret_cast<const float4*>(er + 4 * q);
      int kk = r * 256 + c * 16 + 4 * q;
      actL[kk + 0] = f2bf(fmaxf(av.x + ev.x, 0.f));
      actL[kk + 1] = f2bf(fmaxf(av.y + ev.y, 0.f));
      actL[kk + 2] = f2bf(fmaxf(av.z + ev.z, 0.f));
      actL[kk + 3] = f2bf(fmaxf(av.w + ev.w, 0.f));
    }
  }
  __syncthreads();
  int wave = tid >> 5, lane = tid & 31, h = lane >> 4, r = lane & 15;
  int nt0 = wave * 2;
  v8f acc[2];
  acc[0] = {};
  acc[1] = {};
  const uint16_t* arow = actL + r * 256;
#pragma unroll
  for (int kb = 0; kb < 8; ++kb) {
    const uint16_t* bbase = Ws1 + ((size_t)((kb * 16 + nt0) * 32 + lane) << 4);
    v16bf bb0 = *reinterpret_cast<const v16bf*>(bbase);
    v16bf bb1 = *reinterpret_cast<const v16bf*>(bbase + 512);
    v16bf a   = load_afrag(arow + kb * 32, h);
    acc[0] = wmma_bf16(a, bb0, acc[0]);
    acc[1] = wmma_bf16(a, bb1, acc[1]);
  }
#pragma unroll
  for (int t2 = 0; t2 < 2; ++t2) {
    int col = (nt0 + t2) * 16 + r;
    float bs = b_s1[col], w2 = W_s2[col];
#pragma unroll
    for (int j = 0; j < 8; ++j) {
      float v = fmaxf(acc[t2][j] + bs, 0.f) * w2;
      v += __shfl_xor(v, 1);
      v += __shfl_xor(v, 2);
      v += __shfl_xor(v, 4);
      v += __shfl_xor(v, 8);
      if (r == 0) partL[nt0 + t2][8 * h + j] = v;  // deterministic
    }
  }
  __syncthreads();
  if (tid < 16) {
    float sc = b_s2[0];
#pragma unroll
    for (int nt = 0; nt < 16; ++nt) sc += partL[nt][tid];
    int n = tile * 16 + tid;
    int s = sIdx[n], e = eIdx[n];
    bool valid = (mask[b * T_ + s] != 0) && (mask[b * T_ + e] != 0);
    scores[(size_t)b * NSPAN + n] = valid ? sc : -INFINITY;
  }
}

// ---------------------------------------------------------------------------
// K3: injection GEMM  InjOut[64,256] = inj_emb[64,128] @ W_inj + b_inj
// ---------------------------------------------------------------------------
__global__ void k_inj(const uint16_t* __restrict__ InjBf,
                      const uint16_t* __restrict__ Winj,
                      const float* __restrict__ b_inj,
                      float* __restrict__ InjOut) {
  int lane = threadIdx.x & 31, h = lane >> 4, r = lane & 15;
  int mTile = blockIdx.x, nTile = blockIdx.y;
  v8f acc = {};
  const uint16_t* arow = InjBf + (size_t)(mTile * 16 + r) * J_;
#pragma unroll
  for (int kb = 0; kb < 4; ++kb) {
    v16bf a  = load_afrag(arow + kb * 32, h);
    v16bf bb = *reinterpret_cast<const v16bf*>(
        Winj + ((size_t)((kb * 16 + nTile) * 32 + lane) << 4));
    acc = wmma_bf16(a, bb, acc);
  }
  int col = nTile * 16 + r;
  float bias = b_inj[col];
  float* out = InjOut + (size_t)(mTile * 16 + 8 * h) * H_ + col;
#pragma unroll
  for (int j = 0; j < 8; ++j) out[(size_t)j * H_] = acc[j] + bias;
}

// ---------------------------------------------------------------------------
// K4: exact top-K per batch by rank counting (tie-break = lower index, matches
// lax.top_k), emitted in ascending-index order via chunked prefix compaction.
// ---------------------------------------------------------------------------
__global__ void k_topk(const float* __restrict__ scores,
                       int* __restrict__ topIdx,
                       float* __restrict__ topScore) {
  __shared__ float  sc[NSPAN];     // 33 KB of the WGP's 320 KB LDS
  __shared__ uint8_t fl[NSPAN];
  __shared__ int   csum[256];
  int b = blockIdx.x, tid = threadIdx.x;
  for (int i = tid; i < NSPAN; i += 256) sc[i] = scores[(size_t)b * NSPAN + i];
  __syncthreads();
  const int CH = 33;  // 256*33 >= NSPAN, contiguous chunks keep index order
  int base = tid * CH;
  int cnt = 0;
  for (int ii = 0; ii < CH; ++ii) {
    int i = base + ii;
    if (i >= NSPAN) break;
    float si = sc[i];
    int rank = 0;
    for (int j = 0; j < NSPAN; ++j) {
      float sj = sc[j];
      rank += (sj > si) || (sj == si && j < i);
      if (rank >= KSEL) break;
    }
    uint8_t f = (rank < KSEL) ? 1 : 0;
    fl[i] = f;
    cnt += f;
  }
  csum[tid] = cnt;
  __syncthreads();
  if (tid == 0) {  // small serial exclusive scan, deterministic
    int run = 0;
    for (int t = 0; t < 256; ++t) { int c = csum[t]; csum[t] = run; run += c; }
  }
  __syncthreads();
  int pos = csum[tid];
  for (int ii = 0; ii < CH; ++ii) {
    int i = base + ii;
    if (i >= NSPAN) break;
    if (fl[i]) {
      topIdx[b * KSEL + pos]   = i;
      topScore[b * KSEL + pos] = sc[i];
      ++pos;
    }
  }
}

// ---------------------------------------------------------------------------
// K5: final head over the 256 surviving spans per batch:
//  sec = span_hidden @ W_sec + b_sec ; relu(inj + sec) . W_pred + b_pred
//  + clamped top score -> sigmoid * mask -> out[B,K]
// ---------------------------------------------------------------------------
__global__ void k_head(const float* __restrict__ AE,
                       const int* __restrict__ sIdx,
                       const int* __restrict__ eIdx,
                       const int* __restrict__ mask,
                       const int* __restrict__ topIdx,
                       const float* __restrict__ topScore,
                       const uint16_t* __restrict__ Wsec,
                       const float* __restrict__ b_sec,
                       const float* __restrict__ InjOut,
                       const float* __restrict__ W_pred,
                       const float* __restrict__ b_pred,
                       float* __restrict__ out) {
  __shared__ __align__(16) uint16_t actL[16 * 256];
  __shared__ float partL[16][16];
  __shared__ float maskL[16];
  int tile = blockIdx.x, b = blockIdx.y, tid = threadIdx.x;
  {  // gather span_hidden (no relu here), convert to bf16
    int r = tid >> 4, c = tid & 15;
    int n = topIdx[b * KSEL + tile * 16 + r];
    int s = sIdx[n], e = eIdx[n];
    const float* ar = AE + (size_t)(b * T_ + s) * 512 + c * 16;
    const float* er = AE + (size_t)(b * T_ + e) * 512 + H_ + c * 16;
#pragma unroll
    for (int q = 0; q < 4; ++q) {
      float4 av = *reinterpret_cast<const float4*>(ar + 4 * q);
      float4 ev = *reinterpret_cast<const float4*>(er + 4 * q);
      int kk = r * 256 + c * 16 + 4 * q;
      actL[kk + 0] = f2bf(av.x + ev.x);
      actL[kk + 1] = f2bf(av.y + ev.y);
      actL[kk + 2] = f2bf(av.z + ev.z);
      actL[kk + 3] = f2bf(av.w + ev.w);
    }
    if (c == 0)
      maskL[r] = ((mask[b * T_ + s] != 0) && (mask[b * T_ + e] != 0)) ? 1.f : 0.f;
  }
  __syncthreads();
  int wave = tid >> 5, lane = tid & 31, h = lane >> 4, r = lane & 15;
  int nt0 = wave * 2;
  v8f acc[2];
  acc[0] = {};
  acc[1] = {};
  const uint16_t* arow = actL + r * 256;
#pragma unroll
  for (int kb = 0; kb < 8; ++kb) {
    const uint16_t* bbase = Wsec + ((size_t)((kb * 16 + nt0) * 32 + lane) << 4);
    v16bf bb0 = *reinterpret_cast<const v16bf*>(bbase);
    v16bf bb1 = *reinterpret_cast<const v16bf*>(bbase + 512);
    v16bf a   = load_afrag(arow + kb * 32, h);
    acc[0] = wmma_bf16(a, bb0, acc[0]);
    acc[1] = wmma_bf16(a, bb1, acc[1]);
  }
#pragma unroll
  for (int t2 = 0; t2 < 2; ++t2) {
    int col = (nt0 + t2) * 16 + r;
    float bsec = b_sec[col], wp = W_pred[col];
    float inj = InjOut[b * H_ + col];
#pragma unroll
    for (int j = 0; j < 8; ++j) {
      float v = fmaxf(acc[t2][j] + bsec + inj, 0.f) * wp;
      v += __shfl_xor(v, 1);
      v += __shfl_xor(v, 2);
      v += __shfl_xor(v, 4);
      v += __shfl_xor(v, 8);
      if (r == 0) partL[nt0 + t2][8 * h + j] = v;
    }
  }
  __syncthreads();
  if (tid < 16) {
    float ts = topScore[b * KSEL + tile * 16 + tid];
    if (ts == -INFINITY) ts = -1.0f;
    float lg = b_pred[0] + ts;
#pragma unroll
    for (int nt = 0; nt < 16; ++nt) lg += partL[nt][tid];
    float p = 1.f / (1.f + __expf(-lg));
    out[b * KSEL + tile * 16 + tid] = p * maskL[tid];
  }
}

// ---------------------------------------------------------------------------
// Host-side launcher
// ---------------------------------------------------------------------------
extern "C" void kernel_launch(void* const* d_in, const int* in_sizes, int n_in,
                              void* d_out, int out_size, void* d_ws, size_t ws_size,
                              hipStream_t stream) {
  const float* inputs  = (const float*)d_in[0];
  const int*   in_mask = (const int*)d_in[1];
  const float* inj_emb = (const float*)d_in[2];
  const float* W_start = (const float*)d_in[3];
  const float* b_start = (const float*)d_in[4];
  const float* W_end   = (const float*)d_in[5];
  const float* b_end   = (const float*)d_in[6];
  const float* W_s1    = (const float*)d_in[7];
  const float* b_s1    = (const float*)d_in[8];
  const float* W_s2    = (const float*)d_in[9];
  const float* b_s2    = (const float*)d_in[10];
  const float* W_inj   = (const float*)d_in[11];
  const float* b_inj   = (const float*)d_in[12];
  const float* W_sec   = (const float*)d_in[13];
  const float* b_sec   = (const float*)d_in[14];
  const float* W_pred  = (const float*)d_in[15];
  const float* b_pred  = (const float*)d_in[16];
  float* out = (float*)d_out;

  // workspace carve (~33 MB)
  char* p = (char*)d_ws;
  auto bump = [&](size_t bytes) -> void* {
    void* r = (void*)p;
    p += (bytes + 255) & ~(size_t)255;
    return r;
  };
  uint16_t* Xbf      = (uint16_t*)bump((size_t)BT_ * D_ * 2);
  float*    AE       = (float*)bump((size_t)BT_ * 512 * 4);
  uint16_t* WsePack  = (uint16_t*)bump((size_t)D_ * 512 * 2);
  uint16_t* Ws1Pack  = (uint16_t*)bump((size_t)H_ * H_ * 2);
  uint16_t* WsecPack = (uint16_t*)bump((size_t)H_ * H_ * 2);
  uint16_t* WinjPack = (uint16_t*)bump((size_t)J_ * H_ * 2);
  uint16_t* InjBf    = (uint16_t*)bump((size_t)B_ * J_ * 2);
  float*    InjOut   = (float*)bump((size_t)B_ * H_ * 4);
  float*    scoresW  = (float*)bump((size_t)B_ * NSPAN * 4);
  int*      sIdx     = (int*)bump((size_t)NSPAN * 4);
  int*      eIdx     = (int*)bump((size_t)NSPAN * 4);
  int*      topIdx   = (int*)bump((size_t)B_ * KSEL * 4);
  float*    topScore = (float*)bump((size_t)B_ * KSEL * 4);

  // prep: conversions + fragment packing + span index tables
  {
    int n = BT_ * D_;
    k_cvt_bf16<<<(n + 255) / 256, 256, 0, stream>>>(Xbf, inputs, n);
    n = B_ * J_;
    k_cvt_bf16<<<(n + 255) / 256, 256, 0, stream>>>(InjBf, inj_emb, n);
  }
  {
    int tot = (D_ / 32) * (512 / 16) * 512;  // 393216
    k_pack_w<<<(tot + 255) / 256, 256, 0, stream>>>(WsePack, W_start, W_end, 32, 256, tot);
    tot = (H_ / 32) * (H_ / 16) * 512;       // 65536
    k_pack_w<<<(tot + 255) / 256, 256, 0, stream>>>(Ws1Pack, W_s1, W_s1, 16, 256, tot);
    k_pack_w<<<(tot + 255) / 256, 256, 0, stream>>>(WsecPack, W_sec, W_sec, 16, 256, tot);
    tot = (J_ / 32) * (H_ / 16) * 512;       // 32768
    k_pack_w<<<(tot + 255) / 256, 256, 0, stream>>>(WinjPack, W_inj, W_inj, 16, 256, tot);
  }
  k_span_idx<<<(NSPAN + 255) / 256, 256, 0, stream>>>(sIdx, eIdx);

  // main pipeline
  k_proj<<<dim3(BT_ / 16), 128, 0, stream>>>(Xbf, WsePack, b_start, b_end, AE);
  k_scorer<<<dim3(NTILE_SPAN, B_), 256, 0, stream>>>(AE, sIdx, eIdx, in_mask,
                                                     Ws1Pack, b_s1, W_s2, b_s2, scoresW);
  k_inj<<<dim3(B_ / 16, H_ / 16), 32, 0, stream>>>(InjBf, WinjPack, b_inj, InjOut);
  k_topk<<<B_, 256, 0, stream>>>(scoresW, topIdx, topScore);
  k_head<<<dim3(KSEL / 16, B_), 256, 0, stream>>>(AE, sIdx, eIdx, in_mask,
                                                  topIdx, topScore, WsecPack, b_sec,
                                                  InjOut, W_pred, b_pred, out);
}
